// SequenceModelPadded_62818191671711
// MI455X (gfx1250) — compile-verified, matrix operand order checked by reading, not run on hardware
//
#include <hip/hip_runtime.h>

// Problem dims (fixed by the reference).
#define Bsz  256
#define Tlen 256
#define Idim 512
#define Hdim 1024

typedef __bf16 bf16_t;
typedef bf16_t v16bf __attribute__((ext_vector_type(16)));
typedef float  v8f   __attribute__((ext_vector_type(8)));

#define TILE_M 32           // batch rows per block
#define TILE_N 64           // hidden cols per block
#define TILE_K 64           // K depth per LDS stage (2 WMMAs deep)
#define LDS_STRIDE 72       // ushorts per LDS row (144B, padded vs 64 banks)
#define NBLOCKS ((Hdim / TILE_N) * (Bsz / TILE_M))   // 16*8 = 128

// Pack two f32 -> two bf16 (round-to-nearest) in one v_perm_b32.
__device__ __forceinline__ unsigned pack_bf16_rn(float x, float y) {
  unsigned ux = __builtin_bit_cast(unsigned, x) + 0x8000u;
  unsigned uy = __builtin_bit_cast(unsigned, y) + 0x8000u;
  // result bytes: [0,1] = ux bytes [2,3] (sel 2,3); [2,3] = uy bytes [2,3] (sel 6,7)
  return __builtin_amdgcn_perm(uy, ux, 0x07060302u);
}

struct Frag32B { uint4 lo, hi; };

// 256 threads load a 32x64 f32 tile, convert to bf16, store to LDS.
// Per thread: 2x float4 (8 consecutive floats), one uint4 LDS store.
__device__ __forceinline__ void fill_tile_32x64(const float* __restrict__ src,
                                                int rowStride,
                                                unsigned short* lds) {
  const int tid = threadIdx.x;    // 0..255
  const int row = tid >> 3;       // 0..31
  const int oct = tid & 7;        // 0..7 -> 8 consecutive floats
  const float4* g = reinterpret_cast<const float4*>(src + (size_t)row * rowStride + oct * 8);
  float4 a = g[0];
  float4 b = g[1];
  uint4 o;
  o.x = pack_bf16_rn(a.x, a.y);
  o.y = pack_bf16_rn(a.z, a.w);
  o.z = pack_bf16_rn(b.x, b.y);
  o.w = pack_bf16_rn(b.z, b.w);
  *reinterpret_cast<uint4*>(lds + row * LDS_STRIDE + oct * 8) = o;
}

// 256 threads load a 64x64 f32 tile (two 32-row halves).
__device__ __forceinline__ void fill_tile_64x64(const float* __restrict__ src,
                                                int rowStride,
                                                unsigned short* lds) {
  const int tid = threadIdx.x;
  const int oct = tid & 7;
#pragma unroll
  for (int rr = 0; rr < 2; ++rr) {
    const int row = (tid >> 3) + rr * 32;   // 0..63
    const float4* g = reinterpret_cast<const float4*>(src + (size_t)row * rowStride + oct * 8);
    float4 a = g[0];
    float4 b = g[1];
    uint4 o;
    o.x = pack_bf16_rn(a.x, a.y);
    o.y = pack_bf16_rn(a.z, a.w);
    o.z = pack_bf16_rn(b.x, b.y);
    o.w = pack_bf16_rn(b.z, b.w);
    *reinterpret_cast<uint4*>(lds + row * LDS_STRIDE + oct * 8) = o;
  }
}

// Per-lane 16x32 bf16 fragment per CDNA5 16-bit A/B layout:
// lanes 0-15: row = base+lane,    K-chunks {0..7, 16..23}
// lanes16-31: row = base+lane-16, K-chunks {8..15,24..31}
__device__ __forceinline__ v16bf load_frag(const unsigned short* lds, int rowBase,
                                           int lane, int colBase) {
  const int row = rowBase + (lane & 15);
  const int c0  = colBase + ((lane & 16) ? 8 : 0);
  Frag32B f;
  f.lo = *reinterpret_cast<const uint4*>(lds + row * LDS_STRIDE + c0);
  f.hi = *reinterpret_cast<const uint4*>(lds + row * LDS_STRIDE + c0 + 16);
  return __builtin_bit_cast(v16bf, f);
}

__global__ void init_kernel(float* h0, unsigned* ctr) {
  size_t i = (size_t)blockIdx.x * blockDim.x + threadIdx.x;
  if (i < (size_t)Bsz * Hdim) h0[i] = 0.0f;
  if (i == 0) *ctr = 0u;
}

// Persistent kernel: all T timesteps in one launch, device-wide barrier per step.
__global__ void __launch_bounds__(256)
rnn_persistent_kernel(const float* __restrict__ x,        // (B,T,I)
                      const long long* __restrict__ slen, // (B)
                      const float* __restrict__ Wx,       // (T,H,I)
                      const float* __restrict__ Wh,       // (T,H,H)
                      const float* __restrict__ bias,     // (T,H)
                      float* h0, float* h1,
                      unsigned* barrier_counter) {
  __shared__ __align__(16) unsigned short ldsA[TILE_M * LDS_STRIDE];  // 32x64 bf16
  __shared__ __align__(16) unsigned short ldsB[TILE_N * LDS_STRIDE];  // 64x64 bf16

  const int nBase = blockIdx.x * TILE_N;   // hidden dim
  const int bBase = blockIdx.y * TILE_M;   // batch dim
  const int tid   = threadIdx.x;
  const int lane  = tid & 31;
  const int wave  = tid >> 5;              // 0..7
  const int mOff  = (wave & 1) * 16;       // wave's 16x16 C tile
  const int nOff  = (wave >> 1) * 16;      // 0..48

  float* hPrev = h0;
  float* hNext = h1;

  for (int t = 0; t < Tlen; ++t) {
    v8f acc = {};
    const float* WxT = Wx + (size_t)t * Hdim * Idim;
    const float* WhT = Wh + (size_t)t * Hdim * Hdim;

    // ---- phase 1: X_t @ Wx_t^T ----
    for (int kb = 0; kb < Idim; kb += TILE_K) {
      fill_tile_32x64(x + ((size_t)bBase * Tlen + t) * Idim + kb, Tlen * Idim, ldsA);
      fill_tile_64x64(WxT + (size_t)nBase * Idim + kb, Idim, ldsB);
      if (kb + TILE_K < Idim)
        __builtin_prefetch(WxT + (size_t)nBase * Idim + kb + TILE_K, 0, 1);
      __syncthreads();
#pragma unroll
      for (int kh = 0; kh < 2; ++kh) {
        v16bf a = load_frag(ldsA, mOff, lane, kh * 32);
        v16bf w = load_frag(ldsB, nOff, lane, kh * 32);
        acc = __builtin_amdgcn_wmma_f32_16x16x32_bf16(false, a, false, w,
                                                      (short)0, acc, false, false);
      }
      __syncthreads();
    }
    // ---- phase 2: h_{t-1} @ Wh_t^T (h == 0 at t==0, buffer pre-zeroed) ----
    if (t > 0) {
      for (int kb = 0; kb < Hdim; kb += TILE_K) {
        fill_tile_32x64(hPrev + (size_t)bBase * Hdim + kb, Hdim, ldsA);
        fill_tile_64x64(WhT + (size_t)nBase * Hdim + kb, Hdim, ldsB);
        if (kb + TILE_K < Hdim)
          __builtin_prefetch(WhT + (size_t)nBase * Hdim + kb + TILE_K, 0, 1);
        __syncthreads();
#pragma unroll
        for (int kh = 0; kh < 2; ++kh) {
          v16bf a = load_frag(ldsA, mOff, lane, kh * 32);
          v16bf w = load_frag(ldsB, nOff, lane, kh * 32);
          acc = __builtin_amdgcn_wmma_f32_16x16x32_bf16(false, a, false, w,
                                                        (short)0, acc, false, false);
        }
        __syncthreads();
      }
    }

    // ---- epilogue: bias, tanh, length mask ----
    // C/D layout: lane n<16 -> col n, VGPR r -> row r; lanes 16-31 -> rows 8..15.
    const int col = nBase + nOff + (lane & 15);
    const float bt = bias[t * Hdim + col];
    const int rAdd = (lane & 16) ? 8 : 0;
#pragma unroll
    for (int r = 0; r < 8; ++r) {
      const int row = bBase + mOff + r + rAdd;
      const size_t idx = (size_t)row * Hdim + col;
      const float hp = hPrev[idx];
      const float nv = tanhf(acc[r] + bt);
      const bool  m  = (long long)t < slen[row];
      hNext[idx] = m ? nv : hp;
    }

    // swap state buffers
    float* tmp = hPrev; hPrev = hNext; hNext = tmp;

    // ---- device-wide barrier (monotone counter, no reset races) ----
    if (t < Tlen - 1) {
      __syncthreads();
      if (tid == 0) {
        __threadfence();                         // release h writes
        atomicAdd(barrier_counter, 1u);
        const unsigned target = (unsigned)(t + 1) * (unsigned)NBLOCKS;
        while (atomicAdd(barrier_counter, 0u) < target) {
          __builtin_amdgcn_s_sleep(1);
        }
        __threadfence();                         // acquire other blocks' writes
      }
      __syncthreads();
    }
  }
}

// out[b] = h_final[b,:] . W_out[0,:] + b_out[0]
__global__ void __launch_bounds__(256)
out_kernel(const float* __restrict__ hFin, const float* __restrict__ Wout,
           const float* __restrict__ bout, float* __restrict__ out) {
  __shared__ float red[256];
  const int b = blockIdx.x;
  const int tid = threadIdx.x;
  float s = 0.0f;
  for (int h = tid; h < Hdim; h += 256)
    s += hFin[(size_t)b * Hdim + h] * Wout[h];
  red[tid] = s;
  __syncthreads();
  for (int off = 128; off > 0; off >>= 1) {
    if (tid < off) red[tid] += red[tid + off];
    __syncthreads();
  }
  if (tid == 0) out[b] = red[0] + bout[0];
}

extern "C" void kernel_launch(void* const* d_in, const int* in_sizes, int n_in,
                              void* d_out, int out_size, void* d_ws, size_t ws_size,
                              hipStream_t stream) {
  const float*     xin  = (const float*)d_in[0];     // (B,T,I) f32
  const long long* slen = (const long long*)d_in[1]; // (B) i64
  const float*     Wx   = (const float*)d_in[2];     // (T,H,I)
  const float*     Wh   = (const float*)d_in[3];     // (T,H,H)
  const float*     bias = (const float*)d_in[4];     // (T,H)
  const float*     Wout = (const float*)d_in[5];     // (1,H)
  const float*     bout = (const float*)d_in[6];     // (1,)
  float*           out  = (float*)d_out;             // (B,1)

  // workspace: [counter | h0 (B*H f32) | h1 (B*H f32)]
  unsigned* ctr = (unsigned*)d_ws;
  float* h0 = (float*)((char*)d_ws + 256);
  float* h1 = h0 + (size_t)Bsz * Hdim;

  init_kernel<<<((size_t)Bsz * Hdim + 255) / 256, 256, 0, stream>>>(h0, ctr);

  dim3 grid(Hdim / TILE_N, Bsz / TILE_M);   // 16 x 8 = 128 blocks
  rnn_persistent_kernel<<<grid, 256, 0, stream>>>(xin, slen, Wx, Wh, bias, h0, h1, ctr);

  // T = 256 swaps (even) -> final state back in h0
  out_kernel<<<Bsz, 256, 0, stream>>>(h0, Wout, bout, out);
}
